// hankel_density_76347338654317
// MI455X (gfx1250) — matrix operand order repaired and checked by reading, not compiled, not run on hardware
//
#include <hip/hip_runtime.h>

typedef __bf16 bf16_t;
typedef __attribute__((ext_vector_type(16))) __bf16 v16bf;
typedef __attribute__((ext_vector_type(8)))  float  v8f;

#define LOG_SQRT_2PI 0.9189385332046727f

// ---- CDNA5 async Global->LDS path (gate on builtin availability) ----------
#if __has_builtin(__builtin_amdgcn_global_load_async_to_lds_b128) && \
    __has_builtin(__builtin_amdgcn_s_wait_asynccnt)
#define USE_ASYNC_LDS 1
#else
#define USE_ASYNC_LDS 0
#endif

#if USE_ASYNC_LDS
typedef int v4i_vs __attribute__((vector_size(4 * sizeof(int))));
typedef __attribute__((address_space(1))) v4i_vs* gv4i_p;   // global int4*
typedef __attribute__((address_space(3))) v4i_vs* lv4i_p;   // LDS int4*
__device__ __forceinline__ void async_copy_b128(const void* g, void* l) {
  // LDS offset = low 32 bits of the generic shared-aperture address (ISA 10.2)
  __builtin_amdgcn_global_load_async_to_lds_b128(
      (gv4i_p)(uintptr_t)g, (lv4i_p)(unsigned)(uintptr_t)l, 0, 0);
}
#endif

// ---------------------------------------------------------------------------
// K1: cores f32 [15][128][128][128] (s,i,d,j) -> coresT bf16 stored (s,i,j,d)
// ---------------------------------------------------------------------------
__global__ __launch_bounds__(256)
void convert_cores_kernel(const float* __restrict__ cores,
                          bf16_t* __restrict__ coresT)
{
  __shared__ bf16_t t[128 * 130];
  const int s = blockIdx.x >> 7;
  const int i = blockIdx.x & 127;
  const float* src = cores + ((size_t)s * 128 + i) * 16384;
  bf16_t* dst = coresT + ((size_t)s * 128 + i) * 16384;

  for (int idx = threadIdx.x; idx < 16384; idx += 256) {
    int dd = idx >> 7, j = idx & 127;
    t[dd * 130 + j] = (bf16_t)src[idx];
  }
  __syncthreads();
  for (int idx = threadIdx.x; idx < 16384; idx += 256) {
    int j = idx >> 7, dd = idx & 127;
    dst[(size_t)j * 128 + dd] = t[dd * 130 + j];
  }
}

// ---------------------------------------------------------------------------
// K2: encoder  e[s][n][d] = tanh(relu(x_{n,s} * W1 + b1) @ W2 + b2)  (bf16)
// ---------------------------------------------------------------------------
__global__ __launch_bounds__(256)
void encoder_kernel(const float* __restrict__ X,
                    const float* __restrict__ W1, const float* __restrict__ b1,
                    const float* __restrict__ W2, const float* __restrict__ b2,
                    bf16_t* __restrict__ eAll, int N)
{
  __shared__ float h1[16 * 128];
  const int nb = N >> 4;
  const int s = blockIdx.x / nb;
  const long n0 = (long)(blockIdx.x % nb) * 16;

  for (int idx = threadIdx.x; idx < 2048; idx += 256) {
    int row = idx >> 7, k = idx & 127;
    float x = X[(n0 + row) * 16 + s];
    h1[idx] = fmaxf(x * W1[k] + b1[k], 0.f);
  }
  __syncthreads();
  for (int idx = threadIdx.x; idx < 2048; idx += 256) {
    int row = idx >> 7, d = idx & 127;
    float a = b2[d];
    const float* hr = &h1[row << 7];
#pragma unroll 8
    for (int k = 0; k < 128; ++k) a += hr[k] * W2[k * 128 + d];
    eAll[((size_t)s * N + n0 + row) * 128 + d] = (bf16_t)tanhf(a);
  }
}

// ---------------------------------------------------------------------------
// K3: acc[n] = phi(x_{n,0}, init_w);  tmp0[n][:] = init_w
// ---------------------------------------------------------------------------
__global__ __launch_bounds__(256)
void phi0_kernel(const float* __restrict__ X, const float* __restrict__ init_w,
                 const float* __restrict__ W_nade, const float* __restrict__ b_nade,
                 const float* __restrict__ W_mu,  const float* __restrict__ b_mu,
                 const float* __restrict__ W_sig, const float* __restrict__ b_sig,
                 const float* __restrict__ W_al,  const float* __restrict__ b_al,
                 float* __restrict__ acc, float* __restrict__ tmp0, int N)
{
  __shared__ float th[128], hn[128], prm[30];
  const int tid = threadIdx.x;

  if (tid < 128) th[tid] = tanhf(init_w[tid]);
  __syncthreads();
  if (tid < 128) {
    float a = b_nade[tid];
    for (int i = 0; i < 128; ++i) a += th[i] * W_nade[i * 128 + tid];
    hn[tid] = fmaxf(a, 0.f);
  }
  __syncthreads();
  if (tid < 30) {
    int sel = tid / 10, m = tid % 10;
    const float* W = sel == 0 ? W_mu : (sel == 1 ? W_sig : W_al);
    const float* bb = sel == 0 ? b_mu : (sel == 1 ? b_sig : b_al);
    float a = bb[m];
    for (int k = 0; k < 128; ++k) a += hn[k] * W[k * 10 + m];
    prm[tid] = a;
  }
  __syncthreads();

  const long n = (long)blockIdx.x * 256 + tid;
  const float x = X[n * 16];
  const float* mu = prm, *lg = prm + 10, *al = prm + 20;
  float mx = -1e30f;
  for (int m = 0; m < 10; ++m) mx = fmaxf(mx, al[m]);
  float se = 0.f;
  for (int m = 0; m < 10; ++m) se += expf(al[m] - mx);
  float lseA = mx + logf(se);
  float tvals[10]; float mx2 = -1e30f;
  for (int m = 0; m < 10; ++m) {
    float z = (x - mu[m]) * expf(-lg[m]);
    tvals[m] = al[m] + (-0.5f * z * z - lg[m] - LOG_SQRT_2PI);
    mx2 = fmaxf(mx2, tvals[m]);
  }
  float se2 = 0.f;
  for (int m = 0; m < 10; ++m) se2 += expf(tvals[m] - mx2);
  acc[n] = (mx2 + logf(se2)) - lseA;

  const long base = (long)blockIdx.x * 256 * 128;
  for (int idx = tid; idx < 256 * 128; idx += 256)
    tmp0[base + idx] = init_w[idx & 127];
}

// ---------------------------------------------------------------------------
// K4: one recurrence step (bf16 WMMA GEMM + fused phi)
//   tmp_new[n,j] = sum_{i,d} tmp[n,i]*e[n,d]*core[i,d,j]
//   acc[n]     += phi(x_{n,s+1}, tmp_new[n,:])
// Block: 32 rows x 128 cols, 8 waves = 2(M) x 4(N).
// Dynamic LDS (88 KB): tmp 16K | e 8K | core slice double buffer 2x32K
// Per wave: 2 j-tiles x 2 split-K accumulators -> WMMA dep-chain distance 2.
// ---------------------------------------------------------------------------
#define TT_SMEM_BYTES (16384 + 8192 + 2 * 32768)

__global__ __launch_bounds__(256)
void tt_step_kernel(const float* __restrict__ tmpIn, float* __restrict__ tmpOut,
                    const bf16_t* __restrict__ coreT,   // [128][128][128] (i,j,d)
                    const bf16_t* __restrict__ eS,      // [N][128]
                    const float* __restrict__ X, int s, int N,
                    const float* __restrict__ W_nade, const float* __restrict__ b_nade,
                    const float* __restrict__ W_mu,  const float* __restrict__ b_mu,
                    const float* __restrict__ W_sig, const float* __restrict__ b_sig,
                    const float* __restrict__ W_al,  const float* __restrict__ b_al,
                    float* __restrict__ acc)
{
  extern __shared__ __attribute__((aligned(64))) char smem[];
  float*  ldsTmp = (float*)smem;                     // 32*128 f32  (16 KB)
  bf16_t* ldsE   = (bf16_t*)(smem + 16384);          // 32*128 bf16 ( 8 KB)
  bf16_t* ldsB   = (bf16_t*)(smem + 24576);          // 2 x 128*128 bf16 (64 KB)

  const int tid  = threadIdx.x;
  const int lane = tid & 31, wid = tid >> 5;
  const int wm = wid >> 2;          // 0..1
  const int wn = wid & 3;           // 0..3
  const int hf = lane >> 4;
  const int lrow = lane & 15;
  const long n0 = (long)blockIdx.x * 32;

  // stage tmp rows (f32) and e rows (bf16)
  for (int idx = tid; idx < 32 * 128; idx += 256) {
    int r = idx >> 7, c = idx & 127;
    ldsTmp[idx] = tmpIn[(n0 + r) * 128 + c];
    ldsE[idx]   = eS[(n0 + r) * 128 + c];
  }
  // stage core slice i=0 into buffer 0
#if USE_ASYNC_LDS
#pragma unroll
  for (int t = 0; t < 8; ++t) {
    int idx = tid + 256 * t;
    async_copy_b128(coreT + (size_t)idx * 8, ldsB + (size_t)idx * 8);
  }
  __builtin_amdgcn_s_wait_asynccnt(0);
#else
  {
    const uint4* src = (const uint4*)coreT;
    uint4* dst = (uint4*)ldsB;
    for (int idx = tid; idx < 2048; idx += 256) dst[idx] = src[idx];
  }
#endif
  __syncthreads();

  // e-fragments (contiguous k order, matching B): t -> d0+16*hf+t
  const int myrow = wm * 16 + lrow;
  float ef[4][16];
#pragma unroll
  for (int c = 0; c < 4; ++c)
#pragma unroll
    for (int t = 0; t < 16; ++t) {
      ef[c][t]     = (float)ldsE[myrow * 128 + 32 * c + 16 * hf + t];
    }

  // split-K accumulators: accs[0][jt] for c in {0,1}, accs[1][jt] for c in {2,3}
  v8f accs[2][2] = {};
#if !USE_ASYNC_LDS
  uint4 pre[8];
#endif

  for (int i = 0; i < 128; ++i) {
    const int buf = i & 1;
    bf16_t* Bcur = ldsB + (size_t)buf * 16384;
    bf16_t* Bnxt = ldsB + (size_t)(buf ^ 1) * 16384;

    // start staging slice i+1 (overlaps with the WMMA stream below)
    if (i + 1 < 128) {
      const bf16_t* src = coreT + (size_t)(i + 1) * 16384;
#if USE_ASYNC_LDS
#pragma unroll
      for (int t = 0; t < 8; ++t) {
        int idx = tid + 256 * t;
        async_copy_b128(src + (size_t)idx * 8, Bnxt + (size_t)idx * 8);
      }
#else
#pragma unroll
      for (int t = 0; t < 8; ++t) pre[t] = ((const uint4*)src)[tid + 256 * t];
#endif
    }
    if (i + 2 < 128)
      __builtin_prefetch((const char*)(coreT + (size_t)(i + 2) * 16384) + tid * 128, 0, 1);

    const float ts = ldsTmp[myrow * 128 + i];   // tmp[n,i] scalar (LDS broadcast)

#pragma unroll
    for (int c = 0; c < 4; ++c) {               // d0 = 32*c
      v16bf a;
#pragma unroll
      for (int t = 0; t < 16; ++t) a[t] = (bf16_t)(ef[c][t] * ts);
      v8f* av = accs[c >> 1];                   // alternate accumulator sets per c-pair
#pragma unroll
      for (int jt = 0; jt < 2; ++jt) {
        const int col = wn * 32 + jt * 16 + lrow;
        const v16bf b = *(const v16bf*)(Bcur + col * 128 + 32 * c + 16 * hf);
        av[jt] = __builtin_amdgcn_wmma_f32_16x16x32_bf16(
            false, a, false, b, (short)0, av[jt], false, false);
      }
    }

    if (i + 1 < 128) {
#if USE_ASYNC_LDS
      __builtin_amdgcn_s_wait_asynccnt(0);
#else
      uint4* dst = (uint4*)Bnxt;
#pragma unroll
      for (int t = 0; t < 8; ++t) dst[tid + 256 * t] = pre[t];
#endif
    }
    __syncthreads();   // slice i+1 visible; everyone done reading slice i
  }

  // ---- epilogue: merge split-K, write tmp_new, fused phi ----------------
  float* thv = ldsTmp;                          // reuse (old tmp no longer needed)
#pragma unroll
  for (int jt = 0; jt < 2; ++jt) {
    const v8f v = accs[0][jt] + accs[1][jt];    // merge split-K accumulators
#pragma unroll
    for (int q = 0; q < 8; ++q) {               // C layout: row = 8*hf+q, col = lane%16
      const int m = wm * 16 + 8 * hf + q;
      const int j = wn * 32 + jt * 16 + lrow;
      tmpOut[(n0 + m) * 128 + j] = v[q];
      thv[m * 128 + j] = v[q];
    }
  }
  __syncthreads();

  for (int idx = tid; idx < 32 * 128; idx += 256) thv[idx] = tanhf(thv[idx]);
  __syncthreads();

  float* hn = (float*)ldsB;                     // reuse core-slice LDS
  for (int idx = tid; idx < 32 * 128; idx += 256) {
    int r = idx >> 7, k = idx & 127;
    float a = b_nade[k];
    const float* tr = &thv[r << 7];
#pragma unroll 8
    for (int kk = 0; kk < 128; ++kk) a += tr[kk] * W_nade[kk * 128 + k];
    hn[idx] = fmaxf(a, 0.f);
  }
  __syncthreads();

  float* prm = (float*)ldsE;                    // 32 x 30 f32
  for (int idx = tid; idx < 32 * 30; idx += 256) {
    int r = idx / 30, p = idx % 30, sel = p / 10, m = p % 10;
    const float* W = sel == 0 ? W_mu : (sel == 1 ? W_sig : W_al);
    const float* bb = sel == 0 ? b_mu : (sel == 1 ? b_sig : b_al);
    float a = bb[m];
    const float* hr = &hn[r << 7];
#pragma unroll 8
    for (int k = 0; k < 128; ++k) a += hr[k] * W[k * 10 + m];
    prm[r * 30 + p] = a;
  }
  __syncthreads();

  if (tid < 32) {
    const int r = tid;
    const float x = X[(n0 + r) * 16 + s + 1];
    const float* mu = &prm[r * 30], *lg = mu + 10, *al = mu + 20;
    float mx = -1e30f;
    for (int m = 0; m < 10; ++m) mx = fmaxf(mx, al[m]);
    float se = 0.f;
    for (int m = 0; m < 10; ++m) se += expf(al[m] - mx);
    const float lseA = mx + logf(se);
    float tv[10]; float mx2 = -1e30f;
    for (int m = 0; m < 10; ++m) {
      float z = (x - mu[m]) * expf(-lg[m]);
      tv[m] = al[m] + (-0.5f * z * z - lg[m] - LOG_SQRT_2PI);
      mx2 = fmaxf(mx2, tv[m]);
    }
    float se2 = 0.f;
    for (int m = 0; m < 10; ++m) se2 += expf(tv[m] - mx2);
    acc[n0 + r] += (mx2 + logf(se2)) - lseA;    // unique owner per n
  }
}

// ---------------------------------------------------------------------------
// K5: out = exp(acc)
// ---------------------------------------------------------------------------
__global__ __launch_bounds__(256)
void finalize_kernel(const float* __restrict__ acc, float* __restrict__ out, int N)
{
  int n = blockIdx.x * 256 + threadIdx.x;
  if (n < N) out[n] = expf(acc[n]);
}

// ---------------------------------------------------------------------------
extern "C" void kernel_launch(void* const* d_in, const int* in_sizes, int n_in,
                              void* d_out, int out_size, void* d_ws, size_t ws_size,
                              hipStream_t stream)
{
  const float* X      = (const float*)d_in[0];
  const float* init_w = (const float*)d_in[1];
  const float* cores  = (const float*)d_in[2];
  const float* W_enc1 = (const float*)d_in[3];
  const float* b_enc1 = (const float*)d_in[4];
  const float* W_enc2 = (const float*)d_in[5];
  const float* b_enc2 = (const float*)d_in[6];
  const float* W_nade = (const float*)d_in[7];
  const float* b_nade = (const float*)d_in[8];
  const float* W_mu   = (const float*)d_in[9];
  const float* b_mu   = (const float*)d_in[10];
  const float* W_sig  = (const float*)d_in[11];
  const float* b_sig  = (const float*)d_in[12];
  const float* W_al   = (const float*)d_in[13];
  const float* b_al   = (const float*)d_in[14];
  float* out = (float*)d_out;
  const int N = in_sizes[0] / 16;   // X is [N,1,16]

  char* ws = (char*)d_ws;
  size_t off = 0;
  bf16_t* coresT = (bf16_t*)(ws + off); off += 15ull * 128 * 128 * 128 * 2;
  bf16_t* eAll   = (bf16_t*)(ws + off); off += 15ull * (size_t)N * 128 * 2;
  float*  tmp0   = (float*)(ws + off);  off += (size_t)N * 128 * 4;
  float*  tmp1   = (float*)(ws + off);  off += (size_t)N * 128 * 4;
  float*  acc    = (float*)(ws + off);

  convert_cores_kernel<<<15 * 128, 256, 0, stream>>>(cores, coresT);
  encoder_kernel<<<15 * (N / 16), 256, 0, stream>>>(X, W_enc1, b_enc1, W_enc2, b_enc2, eAll, N);
  phi0_kernel<<<N / 256, 256, 0, stream>>>(X, init_w, W_nade, b_nade,
                                           W_mu, b_mu, W_sig, b_sig, W_al, b_al,
                                           acc, tmp0, N);
  for (int s = 0; s < 15; ++s) {
    const float* ti = (s & 1) ? tmp1 : tmp0;
    float* to = (s & 1) ? tmp0 : tmp1;
    tt_step_kernel<<<N / 32, 256, TT_SMEM_BYTES, stream>>>(
        ti, to, coresT + (size_t)s * 2097152, eAll + (size_t)s * N * 128,
        X, s, N, W_nade, b_nade, W_mu, b_mu, W_sig, b_sig, W_al, b_al, acc);
  }
  finalize_kernel<<<N / 256, 256, 0, stream>>>(acc, out, N);
}
